// GraphNet_1185410973698
// MI455X (gfx1250) — matrix operand order repaired and checked by reading, not compiled, not run on hardware
//
#include <hip/hip_runtime.h>

#define NN 6000
#define HH 128
#define NHEADS 4
#define CATW (HH * (NHEADS + 1))   // 640
#define KBLK 32

typedef __attribute__((ext_vector_type(16))) __bf16 v16bf;
typedef __attribute__((ext_vector_type(8)))  __bf16 v8bf;
typedef __attribute__((ext_vector_type(8)))  float  v8f;
typedef __attribute__((ext_vector_type(4)))  int    v4i;

static __device__ __forceinline__ v16bf join16(v8bf lo, v8bf hi) {
    union { v16bf w; v8bf h[2]; } u;
    u.h[0] = lo; u.h[1] = hi;
    return u.w;
}

// ---------------------------------------------------------------- utilities
__global__ void k_cvt_bf16(const float* __restrict__ in, __bf16* __restrict__ out, int n) {
    int i = blockIdx.x * blockDim.x + threadIdx.x;
    if (i < n) out[i] = (__bf16)in[i];
}

// (K,N) f32 row-major -> (N,K) bf16 "column-major" pack for WMMA B-fragments
__global__ void k_cvt_bf16_t(const float* __restrict__ in, __bf16* __restrict__ out,
                             int K, int N) {
    int i = blockIdx.x * blockDim.x + threadIdx.x;
    if (i >= K * N) return;
    int k = i / N, n = i % N;
    out[(size_t)n * K + k] = (__bf16)in[i];
}

__global__ void k_count_deg(const int* __restrict__ col, int E, float* __restrict__ deg) {
    int i = blockIdx.x * blockDim.x + threadIdx.x;
    if (i < E) atomicAdd(&deg[col[i]], 1.f);
}

__global__ void k_dinv(float* __restrict__ deg, int n) {
    int i = blockIdx.x * blockDim.x + threadIdx.x;
    if (i < n) deg[i] = rsqrtf(deg[i] + 1.f);   // +1 for self loop
}

// layer0: (N,3) @ (3,H)
__global__ void k_lin3(const float* __restrict__ x, const float* __restrict__ W0,
                       float* __restrict__ hw) {
    int i = blockIdx.x * blockDim.x + threadIdx.x;
    if (i >= NN * HH) return;
    int node = i >> 7, j = i & (HH - 1);
    const float* xp = x + node * 3;
    hw[i] = xp[0] * W0[j] + xp[1] * W0[HH + j] + xp[2] * W0[2 * HH + j];
}

// scatter-add messages (edges + self loops)
__global__ void k_scatter(const int* __restrict__ edges, int E,
                          const float* __restrict__ hw, const float* __restrict__ dinv,
                          float* __restrict__ agg) {
    long long i = (long long)blockIdx.x * blockDim.x + threadIdx.x;
    long long total = (long long)(E + NN) * HH;
    if (i >= total) return;
    int e = (int)(i >> 7), j = (int)(i & (HH - 1));
    int r, c; float w;
    if (e < E) { r = edges[e]; c = edges[E + e]; w = dinv[r] * dinv[c]; }
    else       { r = c = e - E; float d = dinv[r]; w = d * d; }
    atomicAdd(&agg[(size_t)c * HH + j], hw[(size_t)r * HH + j] * w);
}

__global__ void k_bias_relu(const float* __restrict__ agg, const float* __restrict__ b,
                            float* __restrict__ out32, __bf16* __restrict__ outbf) {
    int i = blockIdx.x * blockDim.x + threadIdx.x;
    if (i >= NN * HH) return;
    float x = agg[i] + b[i & (HH - 1)];
    x = fmaxf(x, 0.f);
    out32[i] = x;
    outbf[i] = (__bf16)x;
}

__global__ void k_copycol(const float* __restrict__ xr, float* __restrict__ cat) {
    int i = blockIdx.x * blockDim.x + threadIdx.x;
    if (i >= NN * HH) return;
    int node = i >> 7, j = i & (HH - 1);
    cat[(size_t)node * CATW + j] = xr[i];
}

__global__ void k_final(const __bf16* __restrict__ h, const float* __restrict__ Wout,
                        const float* __restrict__ bout, float* __restrict__ out) {
    int i = blockIdx.x * blockDim.x + threadIdx.x;
    if (i >= NN * 3) return;
    int node = i / 3, c = i % 3;
    float s = bout[c];
    const __bf16* hp = h + (size_t)node * HH;
    for (int k = 0; k < HH; ++k) s += (float)hp[k] * Wout[k * 3 + c];
    out[i] = s;
}

// ------------------------------------------------------- generic WMMA GEMM
// C(M,Ncol) = A(M,K) @ B(K,Ncol) [+bias] [relu]; B passed pre-transposed (Ncol,K).
// One 16x16 tile per wave.
__global__ __launch_bounds__(128)
void k_gemm(const __bf16* __restrict__ A, const __bf16* __restrict__ Bt,
            const float* __restrict__ bias, float* __restrict__ out32,
            __bf16* __restrict__ outbf, int M, int K, int Ncol, int relu) {
    int ntile = Ncol >> 4;
    int wid = blockIdx.x * (blockDim.x >> 5) + (threadIdx.x >> 5);
    int total = (M >> 4) * ntile;
    if (wid >= total) return;
    int mt = wid / ntile, nt = wid % ntile;
    int lane = threadIdx.x & 31;
    int lrow = lane & 15, lhi = lane >> 4;

    v8f acc;
    for (int r = 0; r < 8; ++r) acc[r] = 0.f;
    const __bf16* arow = A + (size_t)(mt * 16 + lrow) * K;
    const __bf16* bcol = Bt + (size_t)(nt * 16 + lrow) * K;
    for (int ks = 0; ks < K; ks += 32) {
        v16bf af = join16(*(const v8bf*)(arow + ks + 16 * lhi),
                          *(const v8bf*)(arow + ks + 16 * lhi + 8));
        v16bf bfr = join16(*(const v8bf*)(bcol + ks + 16 * lhi),
                           *(const v8bf*)(bcol + ks + 16 * lhi + 8));
        acc = __builtin_amdgcn_wmma_f32_16x16x32_bf16(false, af, false, bfr,
                                                      (short)0, acc, false, false);
    }
    int n = nt * 16 + lrow;
    float bv = bias ? bias[n] : 0.f;
    float res[8];
    for (int r = 0; r < 8; ++r) {
        float x = acc[r] + bv;
        res[r] = relu ? fmaxf(x, 0.f) : x;
    }
    size_t base = (size_t)(mt * 16 + 8 * lhi) * Ncol + n;
    if (out32) {
        for (int r = 0; r < 8; ++r) out32[base + (size_t)r * Ncol] = res[r];
    }
    if (outbf) {
        for (int r = 0; r < 8; ++r) outbf[base + (size_t)r * Ncol] = (__bf16)res[r];
    }
}

// ------------------------------------------------------ flash attention
// grid: (ceil(375/2), NHEADS), block: 64 (2 waves, 16 queries/wave)
__global__ __launch_bounds__(64)
void k_flash(const __bf16* __restrict__ qall, const __bf16* __restrict__ kall,
             const __bf16* __restrict__ vmat, float* __restrict__ outcat) {
    const int head = blockIdx.y;
    const __bf16* q  = qall + (size_t)head * NN * HH;
    const __bf16* km = kall + (size_t)head * NN * HH;

    __shared__ __bf16 lsK[KBLK][HH];        // keys row-major
    __shared__ __bf16 lsVt[HH][KBLK + 8];   // V transposed (+pad: conflict-free, 16B aligned)
    __shared__ __bf16 lsP[2][16][KBLK];     // per-wave probabilities

    const int tid  = threadIdx.x;
    const int wave = tid >> 5;
    const int lane = tid & 31;
    const int lrow = lane & 15;
    const int lhi  = lane >> 4;

    const int qbase  = (blockIdx.x * 2 + wave) * 16;
    const bool qvalid = (qbase < NN);

    // Q fragments (A-matrix 16x32 bf16 layout), kept resident
    v16bf Qf[4];
    {
        int qr = qbase + lrow; if (qr > NN - 1) qr = NN - 1;
        const __bf16* qp = q + (size_t)qr * HH;
        for (int ks = 0; ks < 4; ++ks)
            Qf[ks] = join16(*(const v8bf*)(qp + ks * 32 + 16 * lhi),
                            *(const v8bf*)(qp + ks * 32 + 16 * lhi + 8));
    }

    v8f O[8];
    for (int t = 0; t < 8; ++t) for (int r = 0; r < 8; ++r) O[t][r] = 0.f;
    float mst[8], lst[8];
    for (int r = 0; r < 8; ++r) { mst[r] = -1e30f; lst[r] = 0.f; }

    const int nkb = (NN + KBLK - 1) / KBLK;
    for (int kb = 0; kb < nkb; ++kb) {
        const int kbase = kb * KBLK;
        __syncthreads();
        // stage K (row-major, async direct-to-LDS when available) and V (transposed)
        for (int i = 0; i < 8; ++i) {
            int lin = (i * 64 + tid) * 8;
            int key = lin >> 7, f = lin & (HH - 1);
            int gk = kbase + key; if (gk > NN - 1) gk = NN - 1;
#if __has_builtin(__builtin_amdgcn_global_load_async_to_lds_b128)
            __builtin_amdgcn_global_load_async_to_lds_b128(
                (__attribute__((address_space(1))) v4i*)(km + (size_t)gk * HH + f),
                (__attribute__((address_space(3))) v4i*)(&lsK[key][f]),
                0, 0);
#else
            *(v8bf*)(&lsK[key][f]) = *(const v8bf*)(km + (size_t)gk * HH + f);
#endif
            v8bf val = *(const v8bf*)(vmat + (size_t)gk * HH + f);
            for (int j = 0; j < 8; ++j) lsVt[f + j][key] = val[j];
        }
        if (kb + 1 < nkb) {
            int pk = kbase + KBLK + (tid >> 1); if (pk > NN - 1) pk = NN - 1;
            __builtin_prefetch(km   + (size_t)pk * HH + (tid & 1) * 64, 0, 0);
            __builtin_prefetch(vmat + (size_t)pk * HH + (tid & 1) * 64, 0, 0);
        }
#if __has_builtin(__builtin_amdgcn_global_load_async_to_lds_b128) && \
    __has_builtin(__builtin_amdgcn_s_wait_asynccnt)
        __builtin_amdgcn_s_wait_asynccnt(0);
#endif
        __syncthreads();

        // S = Q @ K^T for two 16-key tiles
        v8f S[2];
        for (int t = 0; t < 2; ++t) {
            for (int r = 0; r < 8; ++r) S[t][r] = 0.f;
            const __bf16* krow = &lsK[t * 16 + lrow][0];
            for (int ks = 0; ks < 4; ++ks) {
                v16bf Bf = join16(*(const v8bf*)(krow + ks * 32 + 16 * lhi),
                                  *(const v8bf*)(krow + ks * 32 + 16 * lhi + 8));
                S[t] = __builtin_amdgcn_wmma_f32_16x16x32_bf16(false, Qf[ks], false, Bf,
                                                               (short)0, S[t], false, false);
            }
            if (kbase + t * 16 + lrow >= NN)
                for (int r = 0; r < 8; ++r) S[t][r] = -1e30f;
        }

        // online softmax; row stats live in the same lanes as C/D tile rows
        float bm[8];
        for (int r = 0; r < 8; ++r) bm[r] = fmaxf(S[0][r], S[1][r]);
        for (int off = 1; off < 16; off <<= 1)
            for (int r = 0; r < 8; ++r) bm[r] = fmaxf(bm[r], __shfl_xor(bm[r], off, 32));
        float ps[8];
        for (int r = 0; r < 8; ++r) {
            float mnew = fmaxf(mst[r], bm[r]);
            float sc = __expf(mst[r] - mnew);
            mst[r] = mnew;
            float p0 = __expf(S[0][r] - mnew);
            float p1 = __expf(S[1][r] - mnew);
            S[0][r] = p0; S[1][r] = p1;
            ps[r] = p0 + p1;
            lst[r] *= sc;
            for (int t = 0; t < 8; ++t) O[t][r] *= sc;
        }
        for (int off = 1; off < 16; off <<= 1)
            for (int r = 0; r < 8; ++r) ps[r] += __shfl_xor(ps[r], off, 32);
        for (int r = 0; r < 8; ++r) lst[r] += ps[r];

        // re-layout P (C-tile) -> A-fragment via LDS
        for (int t = 0; t < 2; ++t)
            for (int r = 0; r < 8; ++r)
                lsP[wave][8 * lhi + r][t * 16 + lrow] = (__bf16)S[t][r];
        __syncthreads();

        v16bf Pa = join16(*(const v8bf*)(&lsP[wave][lrow][16 * lhi]),
                          *(const v8bf*)(&lsP[wave][lrow][16 * lhi + 8]));
        for (int t = 0; t < 8; ++t) {
            const __bf16* vrow = &lsVt[t * 16 + lrow][0];
            v16bf Bf = join16(*(const v8bf*)(vrow + 16 * lhi),
                              *(const v8bf*)(vrow + 16 * lhi + 8));
            O[t] = __builtin_amdgcn_wmma_f32_16x16x32_bf16(false, Pa, false, Bf,
                                                           (short)0, O[t], false, false);
        }
    }

    if (qvalid) {
        float inv[8];
        for (int r = 0; r < 8; ++r) inv[r] = 1.f / lst[r];
        for (int t = 0; t < 8; ++t) {
            int col = HH + head * HH + t * 16 + lrow;
            for (int r = 0; r < 8; ++r) {
                int row = qbase + 8 * lhi + r;
                outcat[(size_t)row * CATW + col] = O[t][r] * inv[r];
            }
        }
    }
}

// ---------------------------------------------------------------- launcher
extern "C" void kernel_launch(void* const* d_in, const int* in_sizes, int n_in,
                              void* d_out, int out_size, void* d_ws, size_t ws_size,
                              hipStream_t stream) {
    const float* x_rest = (const float*)d_in[0];
    const float* x_rig  = (const float*)d_in[1];
    const int*   e_rest = (const int*)d_in[2];
    const int*   e_rig  = (const int*)d_in[3];
    const float* Wr0 = (const float*)d_in[4];   const float* br0 = (const float*)d_in[5];
    const float* Wr  = (const float*)d_in[6];   const float* br  = (const float*)d_in[7];
    const float* Wg0 = (const float*)d_in[8];   const float* bg0 = (const float*)d_in[9];
    const float* Wg  = (const float*)d_in[10];  const float* bg  = (const float*)d_in[11];
    const float* Wa  = (const float*)d_in[12];  const float* ba  = (const float*)d_in[13];
    const float* Wd0 = (const float*)d_in[14];  const float* bd0 = (const float*)d_in[15];
    const float* Wd  = (const float*)d_in[16];  const float* bd  = (const float*)d_in[17];
    const float* Wout = (const float*)d_in[18]; const float* bout = (const float*)d_in[19];
    float* out = (float*)d_out;
    const int E = in_sizes[2] / 2;

    // workspace carve
    char* p = (char*)d_ws;
    auto alloc = [&](size_t bytes) -> void* {
        void* r = (void*)p; p += (bytes + 255) & ~(size_t)255; return r;
    };
    float*  dinv  = (float*)alloc((size_t)NN * 4);
    float*  hw    = (float*)alloc((size_t)NN * HH * 4);
    float*  agg   = (float*)alloc((size_t)NN * HH * 4);
    float*  hcur  = (float*)alloc((size_t)NN * HH * 4);
    __bf16* hbf   = (__bf16*)alloc((size_t)NN * HH * 2);
    float*  xr32  = (float*)alloc((size_t)NN * HH * 4);
    __bf16* xrbf  = (__bf16*)alloc((size_t)NN * HH * 2);
    __bf16* xgbf  = (__bf16*)alloc((size_t)NN * HH * 2);
    __bf16* qbf   = (__bf16*)alloc((size_t)NHEADS * NN * HH * 2);
    __bf16* kbf   = (__bf16*)alloc((size_t)NHEADS * NN * HH * 2);
    float*  cat32 = (float*)alloc((size_t)NN * CATW * 4);
    __bf16* catbf = (__bf16*)alloc((size_t)NN * CATW * 2);
    __bf16* decA  = (__bf16*)alloc((size_t)NN * HH * 2);
    __bf16* decB  = (__bf16*)alloc((size_t)NN * HH * 2);
    __bf16* Wrbf  = (__bf16*)alloc((size_t)2 * HH * HH * 2);   // transposed per layer
    __bf16* Wgbf  = (__bf16*)alloc((size_t)2 * HH * HH * 2);
    __bf16* Wabf  = (__bf16*)alloc((size_t)NHEADS * HH * HH * 2);
    __bf16* Wd0bf = (__bf16*)alloc((size_t)CATW * HH * 2);
    __bf16* Wdbf  = (__bf16*)alloc((size_t)2 * HH * HH * 2);

    auto cdiv = [](long long a, long long b) { return (unsigned)((a + b - 1) / b); };
    const int NH128 = NN * HH;
    const int WSZ = HH * HH;
    const unsigned gemm_blocks = cdiv((long long)(NN / 16) * (HH / 16), 4);

    // weights -> bf16, pre-transposed to (Ncol, K) for contiguous B-fragments
    for (int i = 0; i < 2; ++i)
        k_cvt_bf16_t<<<cdiv(WSZ, 256), 256, 0, stream>>>(Wr + (size_t)i * WSZ, Wrbf + (size_t)i * WSZ, HH, HH);
    for (int i = 0; i < 2; ++i)
        k_cvt_bf16_t<<<cdiv(WSZ, 256), 256, 0, stream>>>(Wg + (size_t)i * WSZ, Wgbf + (size_t)i * WSZ, HH, HH);
    for (int i = 0; i < NHEADS; ++i)
        k_cvt_bf16_t<<<cdiv(WSZ, 256), 256, 0, stream>>>(Wa + (size_t)i * WSZ, Wabf + (size_t)i * WSZ, HH, HH);
    k_cvt_bf16_t<<<cdiv(CATW * HH, 256), 256, 0, stream>>>(Wd0, Wd0bf, CATW, HH);
    for (int i = 0; i < 2; ++i)
        k_cvt_bf16_t<<<cdiv(WSZ, 256), 256, 0, stream>>>(Wd + (size_t)i * WSZ, Wdbf + (size_t)i * WSZ, HH, HH);

    auto encode = [&](const float* x, const int* edges, const float* W0, const float* b0,
                      const __bf16* Wbf, const float* bs, float* o32, __bf16* obf) {
        (void)hipMemsetAsync(dinv, 0, (size_t)NN * 4, stream);
        k_count_deg<<<cdiv(E, 256), 256, 0, stream>>>(edges + E, E, dinv);
        k_dinv<<<cdiv(NN, 256), 256, 0, stream>>>(dinv, NN);
        k_lin3<<<cdiv(NH128, 256), 256, 0, stream>>>(x, W0, hw);
        (void)hipMemsetAsync(agg, 0, (size_t)NH128 * 4, stream);
        k_scatter<<<cdiv((long long)(E + NN) * HH, 256), 256, 0, stream>>>(edges, E, hw, dinv, agg);
        k_bias_relu<<<cdiv(NH128, 256), 256, 0, stream>>>(agg, b0, hcur, hbf);
        for (int i = 0; i < 2; ++i) {
            k_gemm<<<gemm_blocks, 128, 0, stream>>>(hbf, Wbf + (size_t)i * WSZ,
                                                    nullptr, hw, nullptr, NN, HH, HH, 0);
            (void)hipMemsetAsync(agg, 0, (size_t)NH128 * 4, stream);
            k_scatter<<<cdiv((long long)(E + NN) * HH, 256), 256, 0, stream>>>(edges, E, hw, dinv, agg);
            float* t32 = (i == 1) ? o32 : hcur;
            __bf16* tbf = (i == 1) ? obf : hbf;
            k_bias_relu<<<cdiv(NH128, 256), 256, 0, stream>>>(agg, bs + i * HH, t32, tbf);
        }
    };

    encode(x_rest, e_rest, Wr0, br0, Wrbf, br, xr32, xrbf);
    encode(x_rig,  e_rig,  Wg0, bg0, Wgbf, bg, hw /*discard*/, xgbf);

    // attention: q/k projections then flash
    for (int hd = 0; hd < NHEADS; ++hd) {
        k_gemm<<<gemm_blocks, 128, 0, stream>>>(xrbf, Wabf + (size_t)hd * WSZ,
                                                ba + hd * HH, nullptr,
                                                qbf + (size_t)hd * NN * HH, NN, HH, HH, 0);
        k_gemm<<<gemm_blocks, 128, 0, stream>>>(xgbf, Wabf + (size_t)hd * WSZ,
                                                ba + hd * HH, nullptr,
                                                kbf + (size_t)hd * NN * HH, NN, HH, HH, 0);
    }
    k_copycol<<<cdiv(NH128, 256), 256, 0, stream>>>(xr32, cat32);
    {
        dim3 grid(cdiv(NN / 16, 2), NHEADS);
        k_flash<<<grid, 64, 0, stream>>>(qbf, kbf, xgbf, cat32);
    }

    // decoder
    k_cvt_bf16<<<cdiv((long long)NN * CATW, 256), 256, 0, stream>>>(cat32, catbf, NN * CATW);
    k_gemm<<<gemm_blocks, 128, 0, stream>>>(catbf, Wd0bf, bd0, nullptr, decA, NN, CATW, HH, 1);
    k_gemm<<<gemm_blocks, 128, 0, stream>>>(decA, Wdbf, bd, nullptr, decB, NN, HH, HH, 1);
    k_gemm<<<gemm_blocks, 128, 0, stream>>>(decB, Wdbf + WSZ, bd + HH, nullptr, decA, NN, HH, HH, 1);
    k_final<<<cdiv(NN * 3, 256), 256, 0, stream>>>(decA, Wout, bout, out);
}
